// ConvolutionalSelfAttention_55121610277194
// MI455X (gfx1250) — compile-verified, matrix operand order checked by reading, not run on hardware
//
#include <hip/hip_runtime.h>
#include <cstdint>
#include <cstddef>

// ---------------------------------------------------------------------------
// ConvolutionalSelfAttention on MI455X (gfx1250, wave32, WMMA bf16)
//   B=8, C=512, H=W=64 -> N=4096, D=64
//   queries = Wk@skip (index j), keys = Wq@skip (index i), values = Wv@res
//   softmax over i; out = gamma * V^T P + skip
// ---------------------------------------------------------------------------

#define B_  8
#define C_  512
#define N_  4096
#define D_  64

typedef __attribute__((ext_vector_type(16))) __bf16 v16bf;
typedef __attribute__((ext_vector_type(8)))  float  v8f;

#define WMMA_BF16(A, Bv, Cv) \
  __builtin_amdgcn_wmma_f32_16x16x32_bf16(false, (A), false, (Bv), (short)0, (Cv), false, false)

union BPack { unsigned u; __bf16 h[2]; };
union VPack { uint4 q[2]; v16bf v; };

// 16-element bf16 fragment: elements 0..7 at p, 8..15 at p+16
// (ISA 16-bit A/B layout: per-lane contiguous k-octets).
__device__ __forceinline__ v16bf ld_b16x16(const __bf16* p) {
  VPack u;
  u.q[0] = *(const uint4*)(p);
  u.q[1] = *(const uint4*)(p + 16);
  return u.v;
}

// Same fragment but from fp32 memory, converted to bf16 on the fly.
__device__ __forceinline__ v16bf ld_f32x16_cvt(const float* p) {
  float4 a0 = *(const float4*)(p);
  float4 a1 = *(const float4*)(p + 4);
  float4 a2 = *(const float4*)(p + 16);
  float4 a3 = *(const float4*)(p + 20);
  v16bf A;
  A[0]=(__bf16)a0.x; A[1]=(__bf16)a0.y; A[2]=(__bf16)a0.z; A[3]=(__bf16)a0.w;
  A[4]=(__bf16)a1.x; A[5]=(__bf16)a1.y; A[6]=(__bf16)a1.z; A[7]=(__bf16)a1.w;
  A[8]=(__bf16)a2.x; A[9]=(__bf16)a2.y; A[10]=(__bf16)a2.z; A[11]=(__bf16)a2.w;
  A[12]=(__bf16)a3.x; A[13]=(__bf16)a3.y; A[14]=(__bf16)a3.z; A[15]=(__bf16)a3.w;
  return A;
}

__device__ __forceinline__ void st_pack2(__bf16* dst, float x, float y) {
  BPack p; p.h[0] = (__bf16)x; p.h[1] = (__bf16)y;
  *(unsigned*)dst = p.u;
}

// ---------------------------------------------------------------------------
// Kernel 1: projection GEMM  out = W(COUT x C) @ x(b,:,N), bf16 output.
//   TRANS=true  (Q/K, COUT=64):  out[b][n][COUT]; A=W, B=x; wave w owns tile w.
//   TRANS=false (V,  COUT=512):  out[b][COUT][n]; A=x, B=W; wave owns NT tiles,
//                                x fragment loaded once per k-step, reused NT x.
// All stores are packed b32 (v,v+1 results are address-adjacent in both modes).
// ---------------------------------------------------------------------------
template<int COUT, int NT, bool TRANS>
__global__ void proj_gemm_kernel(const float* __restrict__ x,
                                 const float* __restrict__ Wm,
                                 __bf16* __restrict__ out)
{
  __shared__ __bf16 xs[16][C_];              // 16 KB: x tile, [n][c] bf16
  const int b     = blockIdx.y;
  const int nbase = blockIdx.x * 16;
  const int t     = threadIdx.x;

  { // cooperative stage: coalesced over n (16 lanes -> 64B rows)
    const int nloc = t & 15;
    const int c0   = t >> 4;
    const float* xp = x + (size_t)b * C_ * N_ + nbase + nloc;
    for (int c = c0; c < C_; c += 16)
      xs[nloc][c] = (__bf16)xp[(size_t)c * N_];
  }
  __syncthreads();

  const int wave = t >> 5, lane = t & 31;
  const int lm = lane & 15, lh = lane >> 4;

  if constexpr (TRANS) {
    // ---- Q/K: D tile M=dout, N=spatial; out row-major over dout ----
    const int dt = wave;
    if (dt < (COUT >> 4)) {
      v8f acc = {};
      const float* wrow = Wm + (size_t)(dt * 16 + lm) * C_;   // A row m = lm
      #pragma unroll 4
      for (int ks = 0; ks < C_ / 32; ++ks) {
        const int kb = ks * 32 + lh * 8;
        v16bf A  = ld_f32x16_cvt(wrow + kb);
        v16bf Bv = ld_b16x16(&xs[lm][kb]);                    // B row n = lm
        acc = WMMA_BF16(A, Bv, acc);
      }
      const int n = nbase + lm;
      __bf16* orow = out + ((size_t)b * N_ + n) * COUT + dt * 16 + 8 * lh;
      #pragma unroll
      for (int v = 0; v < 8; v += 2)                          // dout = .. + v
        st_pack2(orow + v, acc[v], acc[v + 1]);
    }
  } else {
    // ---- V: D tile M=spatial, N=cout; out row-major over n ----
    v8f acc[NT] = {};
    #pragma unroll 2
    for (int ks = 0; ks < C_ / 32; ++ks) {
      const int kb = ks * 32 + lh * 8;
      v16bf A = ld_b16x16(&xs[lm][kb]);                       // A row m = lm
      #pragma unroll
      for (int ti = 0; ti < NT; ++ti) {
        const int ct = (wave * NT + ti) * 16 + lm;            // B row n = lm
        v16bf Bv = ld_f32x16_cvt(Wm + (size_t)ct * C_ + kb);
        acc[ti] = WMMA_BF16(A, Bv, acc[ti]);
      }
    }
    #pragma unroll
    for (int ti = 0; ti < NT; ++ti) {
      const int cout = (wave * NT + ti) * 16 + lm;
      __bf16* orow = out + ((size_t)b * COUT + cout) * N_ + nbase + 8 * lh;
      #pragma unroll
      for (int v = 0; v < 8; v += 2)                          // spatial = .. + v
        st_pack2(orow + v, acc[ti][v], acc[ti][v + 1]);
    }
  }
}

// ---------------------------------------------------------------------------
// Kernel 2: fused attention. Block = (batch b, 32 queries). 8 waves.
// LDS: full 32 x 4096 score/probability panel in bf16 (CDNA5's 320KB LDS).
// ---------------------------------------------------------------------------
#define SROW  (N_ + 8)     // padded score row (bf16): rows rotate LDS banks
#define QROW  (D_ + 8)     // padded Q row
#define SS_BYTES   (32 * SROW * 2)
#define QS_BYTES   (32 * QROW * 2)
#define SMEM_BYTES (SS_BYTES + QS_BYTES + 8 * 32 * 4 + 32 * 4)

__global__ void attn_kernel(const __bf16* __restrict__ Qg,   // [B][N][64]
                            const __bf16* __restrict__ Kg,   // [B][N][64]
                            const __bf16* __restrict__ Vt,   // [B][C][N]
                            const float*  __restrict__ skip, // [B][C][N]
                            const float*  __restrict__ gammap,
                            float* __restrict__ out)         // [B][C][N]
{
  extern __shared__ char smem[];
  __bf16* Ss   = (__bf16*)smem;                         // [32][SROW]
  __bf16* Qs   = (__bf16*)(smem + SS_BYTES);            // [32][QROW]
  float*  wmax = (float*)(smem + SS_BYTES + QS_BYTES);  // [8][32]
  float*  linv = wmax + 8 * 32;                         // [32]

  const int b     = blockIdx.y;
  const int jbase = blockIdx.x * 32;
  const int t     = threadIdx.x;
  const int wave  = t >> 5, lane = t & 31;
  const int lm    = lane & 15, lh = lane >> 4;
  const float gam = gammap[0];

  { // stage 32x64 query tile (one b128 per thread)
    const int j  = t >> 3;
    const int d8 = (t & 7) * 8;
    *(uint4*)&Qs[j * QROW + d8] =
        *(const uint4*)(Qg + ((size_t)b * N_ + jbase + j) * D_ + d8);
  }
  __syncthreads();

  // ---- Pass 1: S[j][i] = sum_d K[i][d] Q[j][d]; wave w owns i in [512w, 512w+512)
  float jm0 = -3.0e38f, jm1 = -3.0e38f;
  for (int it = 0; it < 32; ++it) {
    const int ibase = wave * 512 + it * 16;
    const __bf16* krow = Kg + ((size_t)b * N_ + ibase + lm) * D_;  // A row m = lm
    v8f acc0 = {}, acc1 = {};
    #pragma unroll
    for (int ks = 0; ks < 2; ++ks) {
      const int kb = ks * 32 + lh * 8;
      v16bf A  = ld_b16x16(krow + kb);
      v16bf B0 = ld_b16x16(&Qs[lm * QROW + kb]);          // B row n = lm (jt=0)
      v16bf B1 = ld_b16x16(&Qs[(16 + lm) * QROW + kb]);   // jt=1
      acc0 = WMMA_BF16(A, B0, acc0);
      acc1 = WMMA_BF16(A, B1, acc1);
    }
    #pragma unroll
    for (int v = 0; v < 8; v += 2) {           // D: m(i)=v+8*lh, n(j)=lm
      const int irow = ibase + v + 8 * lh;
      st_pack2(&Ss[(size_t)lm * SROW + irow],        acc0[v], acc0[v + 1]);
      st_pack2(&Ss[(size_t)(16 + lm) * SROW + irow], acc1[v], acc1[v + 1]);
      jm0 = fmaxf(jm0, fmaxf(acc0[v], acc0[v + 1]));
      jm1 = fmaxf(jm1, fmaxf(acc1[v], acc1[v + 1]));
    }
  }
  jm0 = fmaxf(jm0, __shfl_xor(jm0, 16, 32));
  jm1 = fmaxf(jm1, __shfl_xor(jm1, 16, 32));
  if (lh == 0) {
    wmax[wave * 32 + lm]      = jm0;
    wmax[wave * 32 + 16 + lm] = jm1;
  }
  __syncthreads();

  // ---- Pass 1b: P = exp(S - max), row sums -> linv[j]. Wave owns 4 j-rows.
  #pragma unroll
  for (int jr = 0; jr < 4; ++jr) {
    const int j = wave * 4 + jr;
    float gmax = -3.0e38f;
    #pragma unroll
    for (int w2 = 0; w2 < 8; ++w2) gmax = fmaxf(gmax, wmax[w2 * 32 + j]);
    float s = 0.f;
    for (int itr = 0; itr < N_ / 64; ++itr) {
      const int i2 = (lane + itr * 32) * 2;
      unsigned* sp = (unsigned*)&Ss[(size_t)j * SROW + i2];
      BPack p; p.u = *sp;
      const float e0 = __expf((float)p.h[0] - gmax);
      const float e1 = __expf((float)p.h[1] - gmax);
      s += e0 + e1;
      p.h[0] = (__bf16)e0; p.h[1] = (__bf16)e1;
      *sp = p.u;
    }
    #pragma unroll
    for (int off = 16; off > 0; off >>= 1) s += __shfl_xor(s, off, 32);
    if (lane == 0) linv[j] = 1.0f / s;
  }
  __syncthreads();

  // ---- Pass 2: O[c][j] = sum_i Vt[c][i] P[i][j]; wave owns 64 channels.
  const int cbase = wave * 64;
  v8f acc[8] = {};                                  // [mt(4)][jt(2)]
  const __bf16* vb = Vt + (size_t)b * C_ * N_;
  #pragma unroll 2
  for (int ks = 0; ks < N_ / 32; ++ks) {
    const int kb = ks * 32 + lh * 8;
    v16bf B0 = ld_b16x16(&Ss[(size_t)lm * SROW + kb]);
    v16bf B1 = ld_b16x16(&Ss[(size_t)(16 + lm) * SROW + kb]);
    #pragma unroll
    for (int mt = 0; mt < 4; ++mt) {
      const __bf16* vrow = vb + (size_t)(cbase + mt * 16 + lm) * N_;
      __builtin_prefetch(vrow + kb + 256, 0, 1);    // global_prefetch_b8: V stream
      v16bf A = ld_b16x16(vrow + kb);
      acc[mt * 2 + 0] = WMMA_BF16(A, B0, acc[mt * 2 + 0]);
      acc[mt * 2 + 1] = WMMA_BF16(A, B1, acc[mt * 2 + 1]);
    }
  }
  // epilogue: gamma / softmax-denominator / residual fused
  const float li0 = linv[lm] * gam;
  const float li1 = linv[16 + lm] * gam;
  #pragma unroll
  for (int mt = 0; mt < 4; ++mt) {
    #pragma unroll
    for (int jt = 0; jt < 2; ++jt) {
      const float li = jt ? li1 : li0;
      const int n = jbase + jt * 16 + lm;
      #pragma unroll
      for (int v = 0; v < 8; ++v) {
        const int ci = cbase + mt * 16 + v + 8 * lh;
        const size_t idx = ((size_t)b * C_ + ci) * N_ + n;
        out[idx] = li * acc[mt * 2 + jt][v] + skip[idx];
      }
    }
  }
}

// ---------------------------------------------------------------------------
extern "C" void kernel_launch(void* const* d_in, const int* in_sizes, int n_in,
                              void* d_out, int out_size, void* d_ws, size_t ws_size,
                              hipStream_t stream)
{
  (void)in_sizes; (void)n_in; (void)out_size; (void)ws_size;
  const float* skip  = (const float*)d_in[0];
  const float* res   = (const float*)d_in[1];
  const float* Wq    = (const float*)d_in[2];
  const float* Wk    = (const float*)d_in[3];
  const float* Wv    = (const float*)d_in[4];
  const float* gamma = (const float*)d_in[5];
  float* out = (float*)d_out;

  // workspace: Q 4MB | K 4MB | V^T 32MB  (bf16)
  char* ws = (char*)d_ws;
  __bf16* Qg = (__bf16*)(ws);
  __bf16* Kg = (__bf16*)(ws + (size_t)B_ * N_ * D_ * 2);
  __bf16* Vt = (__bf16*)(ws + (size_t)2 * B_ * N_ * D_ * 2);

  dim3 pgrid(N_ / 16, B_);
  // queries g = Wk @ skip -> [B][N][64]
  proj_gemm_kernel<D_, 1, true><<<pgrid, 256, 0, stream>>>(skip, Wk, Qg);
  // keys    f = Wq @ skip -> [B][N][64]
  proj_gemm_kernel<D_, 1, true><<<pgrid, 256, 0, stream>>>(skip, Wq, Kg);
  // values hv = Wv @ res  -> [B][C][N] (pre-transposed for pass 2)
  proj_gemm_kernel<C_, 4, false><<<pgrid, 256, 0, stream>>>(res, Wv, Vt);

  dim3 agrid(N_ / 32, B_);
  attn_kernel<<<agrid, 256, SMEM_BYTES, stream>>>(Qg, Kg, Vt, skip, gamma, out);
}
